// AttnBlock_91345364451356
// MI455X (gfx1250) — matrix operand order repaired and checked by reading, compiled, not run on hardware
//
#include <hip/hip_runtime.h>

// ---------------------------------------------------------------------------
// AttnBlock for MI455X (gfx1250): GN -> QKV (WMMA f16) -> flash attention
// (WMMA f16, online softmax, TDM-staged K/V in LDS) -> out-proj + residual.
// All fragment operands are pre-swizzled into the CDNA5 WMMA register layouts
// so every A/B load is a contiguous 32B-per-lane load.
// ---------------------------------------------------------------------------

#define B_DIM 4
#define C_DIM 256
#define N_DIM 4096
#define G_DIM 32

typedef _Float16 half_t;
typedef _Float16 v16h __attribute__((ext_vector_type(16)));
typedef float    v8f  __attribute__((ext_vector_type(8)));
typedef unsigned int u32x4 __attribute__((ext_vector_type(4)));
typedef int          i32x8 __attribute__((ext_vector_type(8)));
typedef int          i32x4 __attribute__((ext_vector_type(4)));

// CDNA5 16-bit A-matrix (16x32) layout: element (m row, k) -> lane*16+elem
__device__ __forceinline__ int a_lane_elem(int m, int k) {
  int lo = k & 1;
  int r2 = (k >> 1) & 15;
  int hs = (r2 >> 2) & 1;                       // which 16-lane half
  int vv = (r2 & 3) | (((r2 >> 3) & 1) << 2);   // VGPR index 0..7
  return (((m & 15) | (hs << 4)) << 4) | ((vv << 1) | lo);
}
// CDNA5 16-bit B-matrix (32x16) layout: element (k, n col) -> lane*16+elem
__device__ __forceinline__ int b_lane_elem(int k, int n) {
  return (((n & 15) | (((k >> 3) & 1) << 4)) << 4) | ((k & 7) | (((k >> 4) & 1) << 3));
}

__device__ __forceinline__ v8f v8f_zero() {
  v8f z;
#pragma unroll
  for (int i = 0; i < 8; ++i) z[i] = 0.0f;
  return z;
}

// 1D Tensor-Data-Mover copy: n8 elements of 8 bytes, global -> LDS.
// D# per CDNA5 ISA ch.8: group0 {count=1, lds_addr, global_addr, type=2},
// group1 {data_size=3(8B), tensor_dim0=n8, tensor_dim1=1, tile_dim0=n8,
//         tile_dim1=1, tensor_dim0_stride=n8}. Groups 2/3 unused (<=2D).
__device__ __forceinline__ void tdm_load_1d(unsigned lds_off, const void* src,
                                            int n8) {
  unsigned long long ga = (unsigned long long)(size_t)src;
  u32x4 g0;
  g0[0] = 1u;                                        // count=1 (valid D#)
  g0[1] = lds_off;                                   // lds_addr (bytes)
  g0[2] = (unsigned)(ga & 0xffffffffull);            // global_addr[31:0]
  g0[3] = (unsigned)((ga >> 32) & 0x01ffffffull) | (2u << 30);  // addr hi,type=2
  i32x8 g1;
  g1[0] = 3 << 16;                                   // data_size = 8B
  g1[1] = (n8 & 0xffff) << 16;                       // tensor_dim0 lo16 @48
  g1[2] = ((n8 >> 16) & 0xffff) | (1 << 16);         // dim0 hi16, tensor_dim1=1
  g1[3] = (n8 & 0xffff) << 16;                       // dim1 hi=0, tile_dim0 @112
  g1[4] = 1;                                         // tile_dim1=1, tile_dim2=0
  g1[5] = n8;                                        // tensor_dim0_stride lo32
  g1[6] = 0;
  g1[7] = 0;
  i32x4 z;
  z[0] = 0; z[1] = 0; z[2] = 0; z[3] = 0;
  i32x8 z8;
#pragma unroll
  for (int i = 0; i < 8; ++i) z8[i] = 0;
  __builtin_amdgcn_tensor_load_to_lds(g0, g1, z, z, z8, 0);
}

// ---------------------------------------------------------------------------
// K0: convert f32 weights to f16 fragments. wq/wk/wv -> A layout, wp -> B^T.
// ---------------------------------------------------------------------------
__global__ __launch_bounds__(256) void wconv_kernel(
    const float* __restrict__ wq, const float* __restrict__ wk,
    const float* __restrict__ wv, const float* __restrict__ wp,
    half_t* __restrict__ wqA, half_t* __restrict__ wkA,
    half_t* __restrict__ wvA, half_t* __restrict__ wpB) {
  int i = blockIdx.x * blockDim.x + threadIdx.x;
  if (i >= C_DIM * C_DIM) return;
  int o = i >> 8;         // out channel (A row)
  int c = i & 255;        // in channel  (k)
  int tile = ((o >> 4) * 8 + (c >> 5)) * 512;
  int ai = tile + a_lane_elem(o & 15, c & 31);
  wqA[ai] = (half_t)wq[i];
  wkA[ai] = (half_t)wk[i];
  wvA[ai] = (half_t)wv[i];
  // wp used transposed: B operand with k=c, col=o
  wpB[tile + b_lane_elem(c & 31, o & 15)] = (half_t)wp[i];
}

// ---------------------------------------------------------------------------
// K1: GroupNorm. One block per (b, g); 8 channels * 4096 = 32768 elements.
// Writes h as f16 in B-fragment swizzled layout: [b][n/16][c/32][32][16].
// ---------------------------------------------------------------------------
__global__ __launch_bounds__(256) void gn_kernel(
    const float* __restrict__ x, const float* __restrict__ gamma,
    const float* __restrict__ beta, half_t* __restrict__ h_sw) {
  __shared__ float red[256], red2[256];
  const int bg = blockIdx.x;
  const int b = bg >> 5, g = bg & 31;
  const int tid = threadIdx.x;
  const float* xb = x + (size_t)(b * C_DIM + g * 8) * N_DIM;

  float s = 0.f, s2 = 0.f;
  for (int i = tid; i < 32768; i += 256) {
    float v = xb[i];
    s += v; s2 += v * v;
  }
  red[tid] = s; red2[tid] = s2;
  __syncthreads();
  for (int st = 128; st > 0; st >>= 1) {
    if (tid < st) { red[tid] += red[tid + st]; red2[tid] += red2[tid + st]; }
    __syncthreads();
  }
  const float mean = red[0] * (1.0f / 32768.0f);
  const float var = red2[0] * (1.0f / 32768.0f) - mean * mean;
  const float inv = __builtin_amdgcn_rsqf(var + 1e-5f);

  for (int i = tid; i < 32768; i += 256) {
    int cl = i >> 12;            // channel within group
    int n = i & 4095;
    int c = g * 8 + cl;
    float v = (xb[i] - mean) * inv * gamma[c] + beta[c];
    int idx = ((b * 256 + (n >> 4)) * 8 + (c >> 5)) * 512 +
              b_lane_elem(c & 31, n & 15);
    h_sw[idx] = (half_t)v;
  }
}

// ---------------------------------------------------------------------------
// K2: QKV projection. One wave per 16(o) x 16(n) output tile, all 3 outputs.
// q scaled by C^-0.5 = 1/16 and stored transposed as A-frags over (n, c).
// k stored as B-frags over (c, m); v stored as B-frags over (m, c).
// ---------------------------------------------------------------------------
__global__ __launch_bounds__(256) void qkv_kernel(
    const half_t* __restrict__ wqA, const half_t* __restrict__ wkA,
    const half_t* __restrict__ wvA,
    const float* __restrict__ bq, const float* __restrict__ bk,
    const float* __restrict__ bv,
    const half_t* __restrict__ h_sw,
    half_t* __restrict__ q_sw, half_t* __restrict__ k_sw,
    half_t* __restrict__ v_sw) {
  const int lane = threadIdx.x & 31;
  const int w = blockIdx.x * 8 + (threadIdx.x >> 5);
  const int nt = w & 255;           // n tile (16 cols)
  const int ot = (w >> 8) & 15;     // o tile (16 rows)
  const int b  = w >> 12;
  const int hf = lane >> 4;
  const int col = lane & 15;

  v8f aq = v8f_zero(), ak = v8f_zero(), av = v8f_zero();
#pragma unroll
  for (int ct = 0; ct < 8; ++ct) {
    const int wt = (ot * 8 + ct) * 512 + lane * 16;
    v16h qa = *(const v16h*)(wqA + wt);
    v16h ka = *(const v16h*)(wkA + wt);
    v16h va = *(const v16h*)(wvA + wt);
    v16h hb = *(const v16h*)(h_sw + ((b * 256 + nt) * 8 + ct) * 512 + lane * 16);
    aq = __builtin_amdgcn_wmma_f32_16x16x32_f16(false, qa, false, hb, (short)0, aq, false, false);
    ak = __builtin_amdgcn_wmma_f32_16x16x32_f16(false, ka, false, hb, (short)0, ak, false, false);
    av = __builtin_amdgcn_wmma_f32_16x16x32_f16(false, va, false, hb, (short)0, av, false, false);
  }
  float bqv[8], bkv[8], bvv[8];
#pragma unroll
  for (int r = 0; r < 8; ++r) {
    int o = ot * 16 + hf * 8 + r;
    bqv[r] = bq[o]; bkv[r] = bk[o]; bvv[r] = bv[o];
  }
  const int qk_base = ((b * 256 + nt) * 8 + (ot >> 1)) * 512;
  const int v_base  = ((b * 128 + (nt >> 1)) * 16 + ot) * 512;
#pragma unroll
  for (int r = 0; r < 8; ++r) {
    int ol = hf * 8 + r;                 // local out-channel (D row)
    int kq = ((ot & 1) << 4) | ol;       // c & 31 within q/k tile
    q_sw[qk_base + a_lane_elem(col, kq)] = (half_t)((aq[r] + bqv[r]) * 0.0625f);
    k_sw[qk_base + b_lane_elem(kq, col)] = (half_t)(ak[r] + bkv[r]);
    v_sw[v_base + b_lane_elem(((nt & 1) << 4) | col, ol)] = (half_t)(av[r] + bvv[r]);
  }
}

// ---------------------------------------------------------------------------
// K3: flash attention. Wave = 16 query rows x full d=256. 64-key blocks.
// The 4 waves of a workgroup share one batch's K/V stream, so each 32KB
// K/V block tile is DMAed once per workgroup into LDS by the Tensor Data
// Mover (double buffered, TENSORcnt-paced), instead of 4x redundant L2 reads.
// S = q^T k (WMMA), online softmax (v_exp + 16-lane shfl_xor reductions),
// P transposed through per-wave LDS into A-frags, O += P V (WMMA).
// ---------------------------------------------------------------------------
__global__ __launch_bounds__(128) void flash_kernel(
    const half_t* __restrict__ q_sw, const half_t* __restrict__ k_sw,
    const half_t* __restrict__ v_sw, half_t* __restrict__ a_sw) {
  __shared__ half_t Kbuf[2][16384] __attribute__((aligned(32)));  // 2 x 32KB
  __shared__ half_t Vbuf[2][16384] __attribute__((aligned(32)));  // 2 x 32KB
  __shared__ half_t Pl[4][16 * 64];                               // 8KB
  const int lane = threadIdx.x & 31;
  const int wl = threadIdx.x >> 5;
  const int w = blockIdx.x * 4 + wl;
  const int nt = w & 255;
  const int b = blockIdx.x >> 6;          // 64 blocks per batch (uniform!)
  const int hf = lane >> 4;
  const int col = lane & 15;
  half_t* P = Pl[wl];

  v16h qf[8];
#pragma unroll
  for (int ct = 0; ct < 8; ++ct)
    qf[ct] = *(const v16h*)(q_sw + ((b * 256 + nt) * 8 + ct) * 512 + lane * 16);

  v8f of[16];
#pragma unroll
  for (int i = 0; i < 16; ++i) of[i] = v8f_zero();
  float m_run[8], l_run[8];
#pragma unroll
  for (int r = 0; r < 8; ++r) { m_run[r] = -3.0e38f; l_run[r] = 0.0f; }

  const half_t* kbB = k_sw + (size_t)b * 256 * 8 * 512;
  const half_t* vbB = v_sw + (size_t)b * 128 * 16 * 512;
  const unsigned koff0 = (unsigned)(size_t)(void*)&Kbuf[0][0];
  const unsigned koff1 = (unsigned)(size_t)(void*)&Kbuf[1][0];
  const unsigned voff0 = (unsigned)(size_t)(void*)&Vbuf[0][0];
  const unsigned voff1 = (unsigned)(size_t)(void*)&Vbuf[1][0];

  // prologue: wave 0 DMAs block 0 into buffer 0
  if (threadIdx.x < 32) {
    tdm_load_1d(koff0, kbB, 4096);
    tdm_load_1d(voff0, vbB, 4096);
  }

  for (int blk = 0; blk < 64; ++blk) {
    const int cur = blk & 1;
    if (threadIdx.x < 32) {
      if (blk + 1 < 64) {   // DMA next block into the other buffer
        tdm_load_1d(cur ? koff0 : koff1, kbB + (blk + 1) * 16384, 4096);
        tdm_load_1d(cur ? voff0 : voff1, vbB + (blk + 1) * 16384, 4096);
        __builtin_amdgcn_s_wait_tensorcnt(2);  // current block's pair is done
      } else {
        __builtin_amdgcn_s_wait_tensorcnt(0);
      }
    }
    __syncthreads();        // publish LDS tiles to all waves
    const half_t* Kc = Kbuf[cur];
    const half_t* Vc = Vbuf[cur];

    v8f sf[4];
#pragma unroll
    for (int j = 0; j < 4; ++j) sf[j] = v8f_zero();
#pragma unroll
    for (int ct = 0; ct < 8; ++ct) {
#pragma unroll
      for (int j = 0; j < 4; ++j) {
        v16h kb = *(const v16h*)(Kc + ((j * 8 + ct) * 32 + lane) * 16);
        sf[j] = __builtin_amdgcn_wmma_f32_16x16x32_f16(false, qf[ct], false, kb, (short)0, sf[j], false, false);
      }
    }
    // ---- online softmax (rows live across the 16 lanes of each half) ----
    float csc[8], rsum[8];
#pragma unroll
    for (int r = 0; r < 8; ++r) {
      float m = fmaxf(fmaxf(sf[0][r], sf[1][r]), fmaxf(sf[2][r], sf[3][r]));
#pragma unroll
      for (int off = 8; off > 0; off >>= 1) m = fmaxf(m, __shfl_xor(m, off, 32));
      float mn = fmaxf(m_run[r], m);
      csc[r] = __builtin_amdgcn_exp2f((m_run[r] - mn) * 1.44269504f);
      m_run[r] = mn;
      rsum[r] = 0.0f;
    }
#pragma unroll
    for (int j = 0; j < 4; ++j)
#pragma unroll
      for (int r = 0; r < 8; ++r) {
        float p = __builtin_amdgcn_exp2f((sf[j][r] - m_run[r]) * 1.44269504f);
        sf[j][r] = p;
        rsum[r] += p;
      }
#pragma unroll
    for (int r = 0; r < 8; ++r) {
      float s = rsum[r];
#pragma unroll
      for (int off = 8; off > 0; off >>= 1) s += __shfl_xor(s, off, 32);
      l_run[r] = l_run[r] * csc[r] + s;
    }
#pragma unroll
    for (int i = 0; i < 16; ++i)
#pragma unroll
      for (int r = 0; r < 8; ++r) of[i][r] *= csc[r];
    // ---- P -> LDS (D-frag order), read back as A-frags ----
#pragma unroll
    for (int j = 0; j < 4; ++j)
#pragma unroll
      for (int r = 0; r < 8; ++r)
        P[(r + 8 * hf) * 64 + j * 16 + col] = (half_t)sf[j][r];
    asm volatile("s_wait_dscnt 0" ::: "memory");
#pragma unroll
    for (int kk = 0; kk < 2; ++kk) {
      v16h pa;
#pragma unroll
      for (int e = 0; e < 16; ++e) {
        int vv = e >> 1, lo = e & 1;
        int kb0 = (vv < 4) ? (2 * vv) : (16 + 2 * (vv - 4));
        pa[e] = P[col * 64 + kk * 32 + kb0 + 8 * hf + lo];
      }
#pragma unroll
      for (int ct = 0; ct < 16; ++ct) {
        v16h vb = *(const v16h*)(Vc + ((kk * 16 + ct) * 32 + lane) * 16);
        of[ct] = __builtin_amdgcn_wmma_f32_16x16x32_f16(false, pa, false, vb, (short)0, of[ct], false, false);
      }
    }
    __syncthreads();        // all waves done reading before buffer reuse
  }
  // ---- epilogue: O /= l, store as A-frags over (n, c) for the out-proj ----
  float invl[8];
#pragma unroll
  for (int r = 0; r < 8; ++r) invl[r] = 1.0f / l_run[r];
  const int abase = ((b * 256 + nt) * 8) * 512;
#pragma unroll
  for (int ct = 0; ct < 16; ++ct)
#pragma unroll
    for (int r = 0; r < 8; ++r) {
      int nrow = r + 8 * hf;
      int kk = ((ct & 1) << 4) | col;
      a_sw[abase + (ct >> 1) * 512 + a_lane_elem(nrow, kk)] =
          (half_t)(of[ct][r] * invl[r]);
    }
}

// ---------------------------------------------------------------------------
// K4: out = x + wp.a + bp. D[n,o] = a^T(n,c) x wp^T(c,o); contiguous stores.
// ---------------------------------------------------------------------------
__global__ __launch_bounds__(256) void proj_kernel(
    const half_t* __restrict__ a_sw, const half_t* __restrict__ wpB,
    const float* __restrict__ bp, const float* __restrict__ x,
    float* __restrict__ out) {
  const int lane = threadIdx.x & 31;
  const int w = blockIdx.x * 8 + (threadIdx.x >> 5);
  const int og = w & 3;             // group of 64 out-channels
  const int nt = (w >> 2) & 255;
  const int b = w >> 10;
  const int hf = lane >> 4;
  const int col = lane & 15;

  v8f ac[4];
#pragma unroll
  for (int jo = 0; jo < 4; ++jo) ac[jo] = v8f_zero();
#pragma unroll
  for (int ct = 0; ct < 8; ++ct) {
    v16h af = *(const v16h*)(a_sw + ((b * 256 + nt) * 8 + ct) * 512 + lane * 16);
#pragma unroll
    for (int jo = 0; jo < 4; ++jo) {
      v16h wb = *(const v16h*)(wpB + ((og * 4 + jo) * 8 + ct) * 512 + lane * 16);
      ac[jo] = __builtin_amdgcn_wmma_f32_16x16x32_f16(false, af, false, wb, (short)0, ac[jo], false, false);
    }
  }
#pragma unroll
  for (int jo = 0; jo < 4; ++jo) {
    int o = (og * 4 + jo) * 16 + col;
    float bb = bp[o];
    size_t base = (size_t)(b * C_DIM + o) * N_DIM + nt * 16 + 8 * hf;
#pragma unroll
    for (int r = 0; r < 8; ++r) {
      size_t idx = base + r;
      out[idx] = x[idx] + ac[jo][r] + bb;
    }
  }
}

// ---------------------------------------------------------------------------
extern "C" void kernel_launch(void* const* d_in, const int* in_sizes, int n_in,
                              void* d_out, int out_size, void* d_ws, size_t ws_size,
                              hipStream_t stream) {
  const float* x  = (const float*)d_in[0];
  const float* gs = (const float*)d_in[1];
  const float* gb = (const float*)d_in[2];
  const float* wq = (const float*)d_in[3];
  const float* bq = (const float*)d_in[4];
  const float* wk = (const float*)d_in[5];
  const float* bk = (const float*)d_in[6];
  const float* wv = (const float*)d_in[7];
  const float* bv = (const float*)d_in[8];
  const float* wp = (const float*)d_in[9];
  const float* bp = (const float*)d_in[10];
  float* out = (float*)d_out;

  // workspace layout (halves): 4 weight frag arrays + h, q^T, k, v^T, a^T
  half_t* wsh = (half_t*)d_ws;
  half_t* wqA = wsh;
  half_t* wkA = wqA + 65536;
  half_t* wvA = wkA + 65536;
  half_t* wpB = wvA + 65536;
  half_t* hsw = wpB + 65536;          // [4][256][8][512]
  half_t* qsw = hsw + 4194304;
  half_t* ksw = qsw + 4194304;
  half_t* vsw = ksw + 4194304;        // [4][128][16][512]
  half_t* asw = vsw + 4194304;

  hipLaunchKernelGGL(wconv_kernel, dim3(256), dim3(256), 0, stream,
                     wq, wk, wv, wp, wqA, wkA, wvA, wpB);
  hipLaunchKernelGGL(gn_kernel, dim3(128), dim3(256), 0, stream,
                     x, gs, gb, hsw);
  hipLaunchKernelGGL(qkv_kernel, dim3(2048), dim3(256), 0, stream,
                     wqA, wkA, wvA, bq, bk, bv, hsw, qsw, ksw, vsw);
  hipLaunchKernelGGL(flash_kernel, dim3(256), dim3(128), 0, stream,
                     qsw, ksw, vsw, asw);
  hipLaunchKernelGGL(proj_kernel, dim3(512), dim3(256), 0, stream,
                     asw, wpB, bp, x, out);
}
